// VGAE_66383014527469
// MI455X (gfx1250) — compile-verified, matrix operand order checked by reading, not run on hardware
//
#include <hip/hip_runtime.h>
#include <hip/hip_bf16.h>
#include <stdint.h>

typedef __bf16 v16bf __attribute__((ext_vector_type(16)));
typedef float  v8f   __attribute__((ext_vector_type(8)));
typedef unsigned int u32x4 __attribute__((ext_vector_type(4)));
typedef float  f32x4 __attribute__((ext_vector_type(4)));

union Frag16 {
    v16bf v;
    u32x4 q[2];
    unsigned short u[16];
};

__device__ __forceinline__ unsigned short f32_bf16(float f) {
    unsigned int x = __float_as_uint(f);
    x += 0x7FFFu + ((x >> 16) & 1u);   // round-to-nearest-even
    return (unsigned short)(x >> 16);
}

// ---------------- elementwise / scatter kernels ----------------

__global__ void k_f32_to_bf16(const float* __restrict__ in,
                              unsigned short* __restrict__ out, size_t n) {
    size_t i = (size_t)blockIdx.x * blockDim.x + threadIdx.x;
    if (i < n) out[i] = f32_bf16(in[i]);
}

__global__ void k_relu_to_bf16(const float* __restrict__ in,
                               unsigned short* __restrict__ out, size_t n) {
    size_t i = (size_t)blockIdx.x * blockDim.x + threadIdx.x;
    if (i < n) out[i] = f32_bf16(fmaxf(in[i], 0.0f));
}

__global__ void k_fill(float* __restrict__ p, float v, int n) {
    int i = blockIdx.x * blockDim.x + threadIdx.x;
    if (i < n) p[i] = v;
}

__global__ void k_deg(const int* __restrict__ col, float* __restrict__ deg, int E) {
    int i = blockIdx.x * blockDim.x + threadIdx.x;
    if (i < E) atomicAdd(&deg[col[i]], 1.0f);
}

__global__ void k_rsqrt(float* __restrict__ d, int n) {
    int i = blockIdx.x * blockDim.x + threadIdx.x;
    if (i < n) d[i] = rsqrtf(d[i]);   // deg >= 1 always (self loop)
}

// out[n][f] = lin[n][f] * dis[n]^2 + bias[f]   (self-loop term + bias, resets buffer)
__global__ void k_init_agg(const float* __restrict__ lin, const float* __restrict__ dis,
                           const float* __restrict__ bias, float* __restrict__ out,
                           int F, size_t total) {
    size_t i = (size_t)blockIdx.x * blockDim.x + threadIdx.x;
    if (i >= total) return;
    size_t node = i / (size_t)F;
    int f = (int)(i - node * (size_t)F);
    float s = dis[node];
    out[i] = lin[i] * s * s + bias[f];
}

// out[col[e]][f] += lin[row[e]][f] * dis[row]*dis[col], 4 feats per thread
__global__ void k_scatter(const float* __restrict__ lin, const float* __restrict__ dis,
                          const int* __restrict__ row, const int* __restrict__ col,
                          float* __restrict__ out, int F, int F4, long long total) {
    long long i = (long long)blockIdx.x * blockDim.x + threadIdx.x;
    if (i >= total) return;
    int e  = (int)(i / F4);
    int f4 = (int)(i - (long long)e * F4);
    int r = row[e], c = col[e];
    float nrm = dis[r] * dis[c];
    const f32x4 v = *(const f32x4*)(lin + (size_t)r * F + f4 * 4);
    float* o = out + (size_t)c * F + f4 * 4;
    atomicAdd(o + 0, v.x * nrm);
    atomicAdd(o + 1, v.y * nrm);
    atomicAdd(o + 2, v.z * nrm);
    atomicAdd(o + 3, v.w * nrm);
}

// ---------------- bf16 WMMA GEMM ----------------
// C[M x Nc] = A[M x K] * B, f32 accumulate.
// TRANSB==0: B is K x Nc row-major (weights). TRANSB==1: B is Nc x K row-major (Z^T case).
// Requires M % 128 == 0, Nc % 128 == 0, K % 64 == 0.
// 256 threads = 8 waves in 4(M) x 2(N) grid; each wave -> 32x64 of the 128x128 C tile.
template <int TRANSB>
__global__ __launch_bounds__(256) void gemm_bf16_wmma(
    const unsigned short* __restrict__ A,
    const unsigned short* __restrict__ B,
    float* __restrict__ C,
    int M, int Nc, int K)
{
    __shared__ unsigned short As[128][72];  // row stride 144B (16B-aligned)
    __shared__ unsigned short Bs[128][72];  // [n][k] layout for both modes

    const int tid   = threadIdx.x;
    const int lane  = tid & 31;
    const int wid   = tid >> 5;
    const int waveM = wid >> 1;   // 0..3
    const int waveN = wid & 1;    // 0..1
    const int lm    = lane & 15;
    const int hi    = lane >> 4;

    const int bM = blockIdx.y * 128;
    const int bN = blockIdx.x * 128;

    v8f acc[2][4];
    const v8f vzero = {0.f, 0.f, 0.f, 0.f, 0.f, 0.f, 0.f, 0.f};
#pragma unroll
    for (int i = 0; i < 2; i++)
#pragma unroll
        for (int j = 0; j < 4; j++) acc[i][j] = vzero;

    for (int k0 = 0; k0 < K; k0 += 64) {
        // ---- A tile 128x64 (row-major in K) ----
#pragma unroll
        for (int q = 0; q < 4; q++) {
            int li = tid + q * 256;            // 0..1023, 8 u32x4 per row
            int r = li >> 3, c = li & 7;
            u32x4 v = *(const u32x4*)(A + (size_t)(bM + r) * K + k0 + c * 8);
            *(u32x4*)&As[r][c * 8] = v;
        }
        // ---- B tile -> Bs[n][k] ----
        if (TRANSB) {
#pragma unroll
            for (int q = 0; q < 4; q++) {
                int li = tid + q * 256;
                int r = li >> 3, c = li & 7;
                u32x4 v = *(const u32x4*)(B + (size_t)(bN + r) * K + k0 + c * 8);
                *(u32x4*)&Bs[r][c * 8] = v;
            }
        } else {
            int n  = tid & 127;
            int kh = (tid >> 7) & 1;
            alignas(16) unsigned short tmp[32];
#pragma unroll
            for (int kk = 0; kk < 32; kk++)
                tmp[kk] = B[(size_t)(k0 + kh * 32 + kk) * Nc + bN + n];
#pragma unroll
            for (int q = 0; q < 4; q++)
                *(u32x4*)&Bs[n][kh * 32 + q * 8] = *(const u32x4*)&tmp[q * 8];
        }
        __syncthreads();

        // ---- WMMA over this K chunk (two K=32 steps) ----
#pragma unroll
        for (int ks = 0; ks < 64; ks += 32) {
            Frag16 a[2], b[4];
#pragma unroll
            for (int i = 0; i < 2; i++) {
                const unsigned short* p = &As[waveM * 32 + i * 16 + lm][0] + ks + hi * 8;
                a[i].q[0] = *(const u32x4*)p;         // K = ks + hi*8 + 0..7
                a[i].q[1] = *(const u32x4*)(p + 16);  // K = ks + 16 + hi*8 + 0..7
            }
#pragma unroll
            for (int j = 0; j < 4; j++) {
                const unsigned short* p = &Bs[waveN * 64 + j * 16 + lm][0] + ks + hi * 8;
                b[j].q[0] = *(const u32x4*)p;
                b[j].q[1] = *(const u32x4*)(p + 16);
            }
#pragma unroll
            for (int i = 0; i < 2; i++)
#pragma unroll
                for (int j = 0; j < 4; j++)
                    acc[i][j] = __builtin_amdgcn_wmma_f32_16x16x32_bf16(
                        false, a[i].v, false, b[j].v,
                        (short)0, acc[i][j], false, false);
        }
        __syncthreads();
    }

    // ---- store C: VGPR v -> M = hi*8 + v; lane&15 -> N ----
#pragma unroll
    for (int i = 0; i < 2; i++) {
        int row0 = bM + waveM * 32 + i * 16 + hi * 8;
#pragma unroll
        for (int j = 0; j < 4; j++) {
            int colc = bN + waveN * 64 + j * 16 + lm;
#pragma unroll
            for (int v = 0; v < 8; v++)
                C[(size_t)(row0 + v) * Nc + colc] = acc[i][j][v];
        }
    }
}

// ---------------- launcher ----------------

extern "C" void kernel_launch(void* const* d_in, const int* in_sizes, int n_in,
                              void* d_out, int out_size, void* d_ws, size_t ws_size,
                              hipStream_t stream) {
    const float* x    = (const float*)d_in[0];
    const int*   ei   = (const int*)d_in[1];     // [2, E]: row then col
    const float* Wenc = (const float*)d_in[2];
    const float* benc = (const float*)d_in[3];
    const float* W1   = (const float*)d_in[4];
    const float* b1   = (const float*)d_in[5];
    const float* W2   = (const float*)d_in[6];
    const float* b2   = (const float*)d_in[7];
    float* out = (float*)d_out;

    const int FIN = 512, H = 256, H2 = 128;
    const int N = in_sizes[0] / FIN;   // 16384
    const int E = in_sizes[1] / 2;     // 524288

    char* ws = (char*)d_ws;
    size_t off = 0;
    auto alloc = [&](size_t bytes) -> void* {
        void* p = ws + off;
        off += (bytes + 255) & ~(size_t)255;
        return p;
    };

    unsigned short* x_bf   = (unsigned short*)alloc((size_t)N * FIN * 2);  // 16 MB
    unsigned short* wenc_b = (unsigned short*)alloc((size_t)FIN * H * 2);
    unsigned short* w1_b   = (unsigned short*)alloc((size_t)H * H2 * 2);
    unsigned short* w2_b   = (unsigned short*)alloc((size_t)H * H2 * 2);
    float*          dis    = (float*)alloc((size_t)N * 4);
    float*          Hlin   = (float*)alloc((size_t)N * H * 4);             // 16 MB
    float*          Hagg   = (float*)alloc((size_t)N * H * 4);             // 16 MB
    unsigned short* h_bf   = (unsigned short*)alloc((size_t)N * H * 2);    // 8 MB
    float*          mulin  = (float*)alloc((size_t)N * H2 * 4);            // 8 MB
    float*          lvlin  = (float*)alloc((size_t)N * H2 * 4);            // 8 MB
    unsigned short* mu_bf  = (unsigned short*)alloc((size_t)N * H2 * 2);   // 4 MB
    (void)ws_size; (void)n_in; (void)out_size;

    float* mu_out = out + (size_t)N * N;
    float* lv_out = mu_out + (size_t)N * H2;

    const int T = 256;
    auto nb = [&](long long n) { return (unsigned)((n + T - 1) / T); };

    // 1) bf16 casts of x and weights
    k_f32_to_bf16<<<nb((long long)N * FIN), T, 0, stream>>>(x, x_bf, (size_t)N * FIN);
    k_f32_to_bf16<<<nb((long long)FIN * H), T, 0, stream>>>(Wenc, wenc_b, (size_t)FIN * H);
    k_f32_to_bf16<<<nb((long long)H * H2), T, 0, stream>>>(W1, w1_b, (size_t)H * H2);
    k_f32_to_bf16<<<nb((long long)H * H2), T, 0, stream>>>(W2, w2_b, (size_t)H * H2);

    // 2) degrees (self loop = 1) -> deg^-1/2
    k_fill<<<nb(N), T, 0, stream>>>(dis, 1.0f, N);
    k_deg<<<nb(E), T, 0, stream>>>(ei + E, dis, E);
    k_rsqrt<<<nb(N), T, 0, stream>>>(dis, N);

    // 3) Hlin = x @ W_enc   (16384 x 512 x 256)
    {
        dim3 g(H / 128, N / 128);
        gemm_bf16_wmma<0><<<g, 256, 0, stream>>>(x_bf, wenc_b, Hlin, N, H, FIN);
    }

    // 4) aggregate + bias, then relu -> bf16
    k_init_agg<<<nb((long long)N * H), T, 0, stream>>>(Hlin, dis, benc, Hagg, H, (size_t)N * H);
    {
        long long tot = (long long)E * (H / 4);
        k_scatter<<<nb(tot), T, 0, stream>>>(Hlin, dis, ei, ei + E, Hagg, H, H / 4, tot);
    }
    k_relu_to_bf16<<<nb((long long)N * H), T, 0, stream>>>(Hagg, h_bf, (size_t)N * H);

    // 5) mu/logvar linear parts (16384 x 256 x 128)
    {
        dim3 g(H2 / 128, N / 128);
        gemm_bf16_wmma<0><<<g, 256, 0, stream>>>(h_bf, w1_b, mulin, N, H2, H);
        gemm_bf16_wmma<0><<<g, 256, 0, stream>>>(h_bf, w2_b, lvlin, N, H2, H);
    }

    // 6) aggregate mu / logvar directly into d_out slices
    k_init_agg<<<nb((long long)N * H2), T, 0, stream>>>(mulin, dis, b1, mu_out, H2, (size_t)N * H2);
    {
        long long tot = (long long)E * (H2 / 4);
        k_scatter<<<nb(tot), T, 0, stream>>>(mulin, dis, ei, ei + E, mu_out, H2, H2 / 4, tot);
    }
    k_init_agg<<<nb((long long)N * H2), T, 0, stream>>>(lvlin, dis, b2, lv_out, H2, (size_t)N * H2);
    {
        long long tot = (long long)E * (H2 / 4);
        k_scatter<<<nb(tot), T, 0, stream>>>(lvlin, dis, ei, ei + E, lv_out, H2, H2 / 4, tot);
    }

    // 7) Adj = mu @ mu^T  (16384 x 16384 x 128) -- store-bandwidth-bound WMMA GEMM
    k_f32_to_bf16<<<nb((long long)N * H2), T, 0, stream>>>(mu_out, mu_bf, (size_t)N * H2);
    {
        dim3 g(N / 128, N / 128);
        gemm_bf16_wmma<1><<<g, 256, 0, stream>>>(mu_bf, mu_bf, out, N, N, H2);
    }
}
